// joint_actor_28690381537988
// MI455X (gfx1250) — compile-verified, matrix operand order checked by reading, not run on hardware
//
#include <hip/hip_runtime.h>
#include <hip/hip_bf16.h>
#include <math.h>

// ---------------- problem constants ----------------
#define NGRAPHS   65536
#define NODES_P   9
#define DIM       64
#define GPB       16                 // graphs per block
#define NPB       (GPB * NODES_P)    // 144 nodes per block
#define HSTR      72                 // LDS h row stride (bf16): 144 B, 16B-aligned, conflict-free
#define WSTR      72                 // LDS weight row stride (bf16), same reasoning
#define NTHREADS  288                // 9 waves of 32; wave w owns node-slot w
#define SOFTPLUS_BIAS 0.5413248546129181f

typedef __attribute__((ext_vector_type(16))) __bf16 v16bf;
typedef __attribute__((ext_vector_type(8)))  __bf16 bf16x8;
typedef __attribute__((ext_vector_type(2)))  __bf16 bf16x2;
typedef __attribute__((ext_vector_type(8)))  float  v8f;
typedef __attribute__((ext_vector_type(4)))  unsigned int uint4v;

union Frag { v16bf v; unsigned int u[8]; uint4v q[2]; };

// ---- conversions / transcendentals (CDNA5 hardware paths, guarded) ----
// RNE f32->bf16 (used for weights, where rounding matters)
__device__ __forceinline__ unsigned short f2bf_rne(float f) {
#if __has_builtin(__builtin_amdgcn_cvt_pk_bf16_f32)
  bf16x2 r = __builtin_amdgcn_cvt_pk_bf16_f32(f, 0.f);
  return (unsigned short)(__builtin_bit_cast(unsigned int, r) & 0xFFFFu);
#else
  unsigned int a = __float_as_uint(f);
  a += 0x7FFFu + ((a >> 16) & 1u);
  return (unsigned short)(a >> 16);
#endif
}
// truncating f32 pair -> packed bf16: single v_perm_b32 ({hi[3],hi[2],lo[3],lo[2]})
__device__ __forceinline__ unsigned int packbf2_trunc(float lo, float hi) {
#if __has_builtin(__builtin_amdgcn_perm)
  return __builtin_amdgcn_perm(__float_as_uint(hi), __float_as_uint(lo), 0x07060302u);
#else
  return (__float_as_uint(lo) >> 16) | (__float_as_uint(hi) & 0xFFFF0000u);
#endif
}
// truncating f32 -> bf16 (activations: tanh-bounded, bf16 WMMA input anyway)
__device__ __forceinline__ unsigned short f2bf_trunc(float f) {
  return (unsigned short)(__float_as_uint(f) >> 16);
}
__device__ __forceinline__ float bf2f_lo(unsigned int p) {
  return __uint_as_float(p << 16);
}
__device__ __forceinline__ float bf2f_hi(unsigned int p) {
  return __uint_as_float(p & 0xFFFF0000u);
}
__device__ __forceinline__ float fast_tanh(float x) {
#if __has_builtin(__builtin_amdgcn_tanhf)
  return __builtin_amdgcn_tanhf(x);                      // v_tanh_f32 (TRANS op)
#else
  return tanhf(x);
#endif
}
// packed bf16 add (v_pk_add_bf16 x4) on a 16-byte chunk
__device__ __forceinline__ uint4v pkadd4(uint4v a, uint4v b) {
  bf16x8 va = __builtin_bit_cast(bf16x8, a);
  bf16x8 vb = __builtin_bit_cast(bf16x8, b);
  bf16x8 vc = va + vb;
  return __builtin_bit_cast(uint4v, vc);
}
__device__ __forceinline__ uint4v ldsq(const unsigned short* p) {
  return *(const uint4v*)p;                              // ds_load_b128 (16B-aligned by layout)
}

__global__ __launch_bounds__(NTHREADS)
void joint_actor_fused(const float* __restrict__ x,
                       const float* __restrict__ W_joint, const float* __restrict__ b_joint,
                       const float* __restrict__ W_torso, const float* __restrict__ b_torso,
                       const float* __restrict__ W_rel1,  const float* __restrict__ b_rel1,
                       const float* __restrict__ W_root1,
                       const float* __restrict__ W_rel2,  const float* __restrict__ b_rel2,
                       const float* __restrict__ W_root2,
                       const float* __restrict__ W_heads, const float* __restrict__ b_heads,
                       float* __restrict__ out)
{
  // ---------------- LDS ----------------
  __shared__ __align__(16) unsigned short sH0[NPB * HSTR];      // 20736 B (ping)
  __shared__ __align__(16) unsigned short sH1[NPB * HSTR];      // 20736 B (pong)
  __shared__ __align__(16) unsigned short sWrel1[DIM * WSTR];   // 9216 B, [n][k] bf16
  __shared__ __align__(16) unsigned short sWroot1[DIM * WSTR];
  __shared__ __align__(16) unsigned short sWrel2[DIM * WSTR];
  __shared__ __align__(16) unsigned short sWroot2[DIM * WSTR];
  __shared__ float sBrel1[DIM], sBrel2[DIM];
  __shared__ float sWt[11 * DIM], sBt[DIM];                     // torso proj (f32)
  __shared__ float sWj[2 * DIM],  sBj[DIM];                     // joint proj (f32)
  __shared__ float sWh[8 * DIM * 2], sBh[16];                   // heads (f32)

  const int tid  = threadIdx.x;
  const int lane = tid & 31;
  const int wv   = tid >> 5;          // wave id == node slot 0..8
  const int m    = lane & 15;         // graph row within 16-row tile
  const int hi   = lane >> 4;         // half-wave selector for K layout

  // ------------- cooperative weight preload (transpose D x D to [n][k] bf16, RNE) -------------
  for (int i = tid; i < DIM * DIM; i += NTHREADS) {
    const int n = i >> 6, k = i & 63;
    const int o = n * WSTR + k;
    sWrel1[o]  = f2bf_rne(W_rel1 [k * DIM + n]);
    sWroot1[o] = f2bf_rne(W_root1[k * DIM + n]);
    sWrel2[o]  = f2bf_rne(W_rel2 [k * DIM + n]);
    sWroot2[o] = f2bf_rne(W_root2[k * DIM + n]);
  }
  for (int i = tid; i < DIM; i += NTHREADS) {
    sBrel1[i] = b_rel1[i];  sBrel2[i] = b_rel2[i];
    sBt[i] = b_torso[i];    sBj[i] = b_joint[i];
  }
  for (int i = tid; i < 11 * DIM; i += NTHREADS) sWt[i] = W_torso[i];
  for (int i = tid; i < 2 * DIM;  i += NTHREADS) sWj[i] = W_joint[i];
  for (int i = tid; i < 8 * DIM * 2; i += NTHREADS) sWh[i] = W_heads[i];
  if (tid < 16) sBh[tid] = b_heads[tid];
  __syncthreads();

  // ------------- stage 0: input projections (tiny K -> VALU), write bf16 h0 -------------
  {
    const int node = tid >> 1;                 // 0..143
    const int half = tid & 1;                  // which 32 features
    const int s    = node % NODES_P;
    const size_t gnode = (size_t)blockIdx.x * NPB + (size_t)node;
    const float* xr = x + gnode * 11;
    if (s == 0) {
      float xv[11];
      #pragma unroll
      for (int k = 0; k < 11; ++k) xv[k] = xr[k];
      #pragma unroll
      for (int f = half * 32; f < half * 32 + 32; f += 2) {
        float a0 = sBt[f], a1 = sBt[f + 1];
        #pragma unroll
        for (int k = 0; k < 11; ++k) {
          a0 += xv[k] * sWt[k * DIM + f];
          a1 += xv[k] * sWt[k * DIM + f + 1];
        }
        *(unsigned int*)&sH0[node * HSTR + f] = packbf2_trunc(a0, a1);
      }
    } else {
      const float x0 = xr[0], x1 = xr[1];
      #pragma unroll
      for (int f = half * 32; f < half * 32 + 32; f += 2) {
        float a0 = sBj[f]     + x0 * sWj[f]     + x1 * sWj[DIM + f];
        float a1 = sBj[f + 1] + x0 * sWj[f + 1] + x1 * sWj[DIM + f + 1];
        *(unsigned int*)&sH0[node * HSTR + f] = packbf2_trunc(a0, a1);
      }
    }
  }
  __syncthreads();

  // ------------- GraphConv layer: hout = tanh(agg@Wrel + b + h@Wroot), via WMMA bf16 -------------
  // A/B fragment layout (ISA 16-bit A): VGPRs 0..3 = K pairs from base kc*32+8*hi
  // (16 contiguous bytes), VGPRs 4..7 = same at +32 bytes -> two ds_load_b128 each.
  auto gconv = [&](const unsigned short* __restrict__ hin,
                   unsigned short* __restrict__ hout,
                   const unsigned short* __restrict__ wrelT,
                   const unsigned short* __restrict__ wrootT,
                   const float* __restrict__ brel) {
    Frag a_h[2], a_agg[2];
    const unsigned short* rowSelf  = hin + (m * NODES_P + wv) * HSTR;
    const unsigned short* rowTorso = hin + (m * NODES_P) * HSTR;
    #pragma unroll
    for (int kc = 0; kc < 2; ++kc) {
      const int b0 = kc * 32 + 8 * hi;         // base K (bf16 elems) for VGPRs 0..3
      a_h[kc].q[0] = ldsq(rowSelf + b0);
      a_h[kc].q[1] = ldsq(rowSelf + b0 + 16);
      if (wv == 0) {                           // torso rows: pk-add the 8 joint rows
        uint4v s0 = ldsq(rowTorso + 1 * HSTR + b0);
        uint4v s1 = ldsq(rowTorso + 1 * HSTR + b0 + 16);
        #pragma unroll
        for (int j = 2; j < 9; ++j) {
          s0 = pkadd4(s0, ldsq(rowTorso + j * HSTR + b0));
          s1 = pkadd4(s1, ldsq(rowTorso + j * HSTR + b0 + 16));
        }
        a_agg[kc].q[0] = s0;
        a_agg[kc].q[1] = s1;
      } else {                                 // joint rows: neighbor is the torso
        a_agg[kc].q[0] = ldsq(rowTorso + b0);
        a_agg[kc].q[1] = ldsq(rowTorso + b0 + 16);
      }
    }
    #pragma unroll
    for (int nt = 0; nt < 4; ++nt) {
      v8f c = {0.f, 0.f, 0.f, 0.f, 0.f, 0.f, 0.f, 0.f};
      const int ncol = nt * 16 + m;            // this lane's output column (C layout: N = lane%16)
      const unsigned short* wr = wrelT  + ncol * WSTR;
      const unsigned short* wo = wrootT + ncol * WSTR;
      #pragma unroll
      for (int kc = 0; kc < 2; ++kc) {
        const int b0 = kc * 32 + 8 * hi;
        Frag brl, brt;
        brl.q[0] = ldsq(wr + b0);  brl.q[1] = ldsq(wr + b0 + 16);
        brt.q[0] = ldsq(wo + b0);  brt.q[1] = ldsq(wo + b0 + 16);
        c = __builtin_amdgcn_wmma_f32_16x16x32_bf16(false, a_agg[kc].v, false, brl.v,
                                                    (short)0, c, false, false);
        c = __builtin_amdgcn_wmma_f32_16x16x32_bf16(false, a_h[kc].v,   false, brt.v,
                                                    (short)0, c, false, false);
      }
      const float bias = brel[ncol];
      #pragma unroll
      for (int r = 0; r < 8; ++r) {            // C layout: row M = r + 8*hi
        const float t = fast_tanh(c[r] + bias);      // v_tanh_f32, no branches
        const int mrow = r + 8 * hi;
        hout[(mrow * NODES_P + wv) * HSTR + ncol] = f2bf_trunc(t);  // 1-op convert
      }
    }
  };

  gconv(sH0, sH1, sWrel1, sWroot1, sBrel1);
  __syncthreads();
  gconv(sH1, sH0, sWrel2, sWroot2, sBrel2);
  __syncthreads();

  // ------------- heads: per-joint Linear(64,2) -> loc, biased-softplus scale -------------
  if (tid < GPB * 8) {
    const int g = tid >> 3, j = tid & 7;
    const unsigned short* hrow = sH0 + (g * NODES_P + 1 + j) * HSTR;
    float a0 = sBh[j * 2 + 0];
    float a1 = sBh[j * 2 + 1];
    #pragma unroll
    for (int k = 0; k < DIM; k += 2) {
      const unsigned int p = *(const unsigned int*)&hrow[k];
      const float h0 = bf2f_lo(p), h1 = bf2f_hi(p);
      a0 += h0 * sWh[(j * DIM + k) * 2 + 0] + h1 * sWh[(j * DIM + k + 1) * 2 + 0];
      a1 += h0 * sWh[(j * DIM + k) * 2 + 1] + h1 * sWh[(j * DIM + k + 1) * 2 + 1];
    }
    const size_t gg = (size_t)blockIdx.x * GPB + (size_t)g;
    out[gg * 8 + j] = a0;                                     // loc
    const float v = a1 + SOFTPLUS_BIAS;
    const float sp = (v > 20.f) ? v : __logf(1.f + __expf(v));  // stable softplus, fast path
    out[(size_t)NGRAPHS * 8 + gg * 8 + j] = sp;               // scale
  }
}

extern "C" void kernel_launch(void* const* d_in, const int* in_sizes, int n_in,
                              void* d_out, int out_size, void* d_ws, size_t ws_size,
                              hipStream_t stream) {
  (void)in_sizes; (void)n_in; (void)out_size; (void)d_ws; (void)ws_size;
  const float* x       = (const float*)d_in[0];
  // d_in[1] = edge_index: fixed bidirectional star, topology hard-coded in kernel
  const float* W_joint = (const float*)d_in[2];
  const float* b_joint = (const float*)d_in[3];
  const float* W_torso = (const float*)d_in[4];
  const float* b_torso = (const float*)d_in[5];
  const float* W_rel1  = (const float*)d_in[6];
  const float* b_rel1  = (const float*)d_in[7];
  const float* W_root1 = (const float*)d_in[8];
  const float* W_rel2  = (const float*)d_in[9];
  const float* b_rel2  = (const float*)d_in[10];
  const float* W_root2 = (const float*)d_in[11];
  const float* W_heads = (const float*)d_in[12];
  const float* b_heads = (const float*)d_in[13];
  float* out = (float*)d_out;

  dim3 grid(NGRAPHS / GPB);      // 4096 blocks
  dim3 block(NTHREADS);          // 9 waves (one per node slot)
  hipLaunchKernelGGL(joint_actor_fused, grid, block, 0, stream,
                     x, W_joint, b_joint, W_torso, b_torso,
                     W_rel1, b_rel1, W_root1,
                     W_rel2, b_rel2, W_root2,
                     W_heads, b_heads, out);
}
